// KAN_Convolutional_Layer_84378927497340
// MI455X (gfx1250) — matrix operand order repaired and checked by reading, not compiled, not run on hardware
//
#include <hip/hip_runtime.h>
#include <hip/hip_bf16.h>

typedef __attribute__((ext_vector_type(16))) _Float16 v16h;
typedef __attribute__((ext_vector_type(8)))  _Float16 v8h;
typedef __attribute__((ext_vector_type(8)))  float    v8f;

namespace {
constexpr int NK    = 9;     // 3x3 taps
constexpr int NCONV = 8;
constexpr int COEFF = 8;     // GRID_SIZE + SPLINE_ORDER
constexpr int IMH = 64, IMW = 64;
constexpr int HO = 62, WO = 62;
constexpr int HOWO = HO * WO;                    // 3844
constexpr int NIMG = 8 * 32;                     // B*C
constexpr int TILES_PER_IMG = (HOWO + 31) / 32;  // 121 (last tile: 4 valid)
constexpr int KDIM = 96;                         // padded feature length (3 WMMA k-steps)
constexpr int ROWSTRIDE = 104;                   // f16 elems; 208B row (16B aligned, bank-skewed)
constexpr int WAVES = 8;                         // 256-thread block = 8 wave32
// feature slot map (== Wcomb row index): [0..8] SiLU, [9..15] zero,
// [16 + k*8 + c] spline bases, [88..95] zero.
}

// ---------------------------------------------------------------------------
// Pack combined (base + scaled spline) weights, f16, directly in the
// v_wmma_f32_16x16x32_f16 B-fragment lane order:
//   bfrag[((s*32 + lane)*16) + e]  with  K = s*32 + (lane>=16)*8 + (e&7) + (e>=8)*16,
//   column = lane & 15  (columns 8..15 padded with zeros).
// ---------------------------------------------------------------------------
__global__ void kan_prep_weights(const float* __restrict__ bw,
                                 const float* __restrict__ sw,
                                 const float* __restrict__ sc,
                                 _Float16* __restrict__ bfrag) {
  for (int idx = blockIdx.x * blockDim.x + threadIdx.x; idx < 3 * 32 * 16;
       idx += gridDim.x * blockDim.x) {
    const int s    = idx >> 9;          // / (32*16)
    const int lane = (idx >> 4) & 31;
    const int e    = idx & 15;
    const int K    = s * 32 + ((lane >= 16) ? 8 : 0) + (e & 7) + ((e >= 8) ? 16 : 0);
    const int v    = lane & 15;
    float val = 0.0f;
    if (v < NCONV) {
      if (K < NK) {
        val = bw[v * NK + K];
      } else if (K >= 16 && K < 16 + NK * COEFF) {
        const int k = (K - 16) >> 3;
        const int c = (K - 16) & 7;
        val = sw[(v * NK + k) * COEFF + c] * sc[v * NK + k];
      }
    }
    bfrag[idx] = (_Float16)val;
  }
}

// ---------------------------------------------------------------------------
// Main fused KAN-conv kernel: one wave32 = one 32-patch tile of one image.
// ---------------------------------------------------------------------------
__launch_bounds__(256)
__global__ void kan_conv_wmma(const float* __restrict__ x,
                              const float* __restrict__ grid,
                              const _Float16* __restrict__ bfrag,
                              float* __restrict__ out) {
  __shared__ __align__(64) _Float16 feat[WAVES][32][ROWSTRIDE]; // 53,248 B
  __shared__ __align__(64) float    outs[WAVES][NCONV][32];     //  8,192 B

  const int lane = threadIdx.x & 31;
  const int wv   = threadIdx.x >> 5;
  const int tile = blockIdx.x * WAVES + wv;          // 30,976 tiles exactly
  const int img  = tile / TILES_PER_IMG;
  const int q0   = (tile - img * TILES_PER_IMG) * 32;
  int q = q0 + lane;
  if (q > HOWO - 1) q = HOWO - 1;                    // clamp: keep EXEC full for WMMA
  const int ho = q / WO;
  const int wo = q - ho * WO;

  // B fragments: 3 k-steps, kept in VGPRs for the whole tile
  const v16h bm0 = *(const v16h*)(bfrag + (0 * 32 + lane) * 16);
  const v16h bm1 = *(const v16h*)(bfrag + (1 * 32 + lane) * 16);
  const v16h bm2 = *(const v16h*)(bfrag + (2 * 32 + lane) * 16);

  // uniform-knot spline parameters (grid is uniform by construction);
  // v_rcp_f32 instead of the IEEE divide expansion — result feeds f16 anyway
  const float g0    = grid[0];
  const float inv_h = __builtin_amdgcn_rcpf(grid[1] - g0);

  // ---- per-lane feature row: zero, then fill SiLU + 4 nonzero bases per tap
  _Float16* row = &feat[wv][lane][0];
  {
    v8h z = {};
#pragma unroll
    for (int i = 0; i < KDIM / 8; ++i) *(v8h*)(row + i * 8) = z;
  }

  const float* xim = x + (size_t)img * (IMH * IMW);
#pragma unroll
  for (int i = 0; i < 3; ++i) {
    const float* xr = xim + (ho + i) * IMW + wo;
#pragma unroll
    for (int j = 0; j < 3; ++j) {
      const int   k  = i * 3 + j;
      const float xv = xr[j];
      // SiLU via v_exp_f32 + v_rcp_f32 (1 ulp; f16 rounding dominates)
      row[k] = (_Float16)(xv * __builtin_amdgcn_rcpf(1.0f + __expf(-xv)));
      // closed-form uniform cubic B-spline: 4 nonzero of 8 bases
      const float f = (xv - g0) * inv_h;
      int cell = (int)floorf(f);
      cell = cell < 3 ? 3 : (cell > 7 ? 7 : cell);
      const float u  = f - (float)cell;
      const float u2 = u * u, u3 = u2 * u, om = 1.0f - u;
      _Float16* bp = row + 16 + k * 8 + (cell - 3);
      bp[0] = (_Float16)(om * om * om * (1.0f / 6.0f));
      bp[1] = (_Float16)((3.0f * u3 - 6.0f * u2 + 4.0f) * (1.0f / 6.0f));
      bp[2] = (_Float16)((-3.0f * u3 + 3.0f * u2 + 3.0f * u + 1.0f) * (1.0f / 6.0f));
      bp[3] = (_Float16)(u3 * (1.0f / 6.0f));
    }
  }

  // cross-lane LDS handoff within the wave (same-wave DS ops are in order)
  asm volatile("s_wait_dscnt 0" ::: "memory");

  // ---- GEMM: features(32x96) x Wcomb(96x16) via 6x v_wmma_f32_16x16x32_f16
  const int r16 = lane & 15;
  const int hi8 = (lane >> 4) << 3;   // upper half-wave supplies the K+8 halves
  v8f acc0 = {}, acc1 = {};
#pragma unroll
  for (int s = 0; s < 3; ++s) {
    const v16h bm = (s == 0) ? bm0 : (s == 1) ? bm1 : bm2;

    const _Float16* r0 = &feat[wv][r16][s * 32 + hi8];
    const v8h a0lo = *(const v8h*)r0;          // ds_load_b128
    const v8h a0hi = *(const v8h*)(r0 + 16);   // ds_load_b128
    const v16h a0 = __builtin_shufflevector(a0lo, a0hi,
        0, 1, 2, 3, 4, 5, 6, 7, 8, 9, 10, 11, 12, 13, 14, 15);
    acc0 = __builtin_amdgcn_wmma_f32_16x16x32_f16(false, a0, false, bm,
                                                  (short)0, acc0, false, false);

    const _Float16* r1 = &feat[wv][16 + r16][s * 32 + hi8];
    const v8h a1lo = *(const v8h*)r1;
    const v8h a1hi = *(const v8h*)(r1 + 16);
    const v16h a1 = __builtin_shufflevector(a1lo, a1hi,
        0, 1, 2, 3, 4, 5, 6, 7, 8, 9, 10, 11, 12, 13, 14, 15);
    acc1 = __builtin_amdgcn_wmma_f32_16x16x32_f16(false, a1, false, bm,
                                                  (short)0, acc1, false, false);
  }

  // ---- D scatter through LDS: lane n holds conv n, patches (hi8..hi8+7)
  if (r16 < NCONV) {
    *(v8f*)(&outs[wv][r16][hi8])      = acc0;  // patches  0..15
    *(v8f*)(&outs[wv][r16][16 + hi8]) = acc1;  // patches 16..31
  }
  asm volatile("s_wait_dscnt 0" ::: "memory");

  // ---- coalesced global stores: 8 x 128B lines per wave
  const bool ok = (q0 + lane) < HOWO;
  float* obase = out + (size_t)(img * NCONV) * HOWO + (q0 + lane);
#pragma unroll
  for (int v = 0; v < NCONV; ++v)
    if (ok) obase[(size_t)v * HOWO] = outs[wv][v][lane];
}

// ---------------------------------------------------------------------------
extern "C" void kernel_launch(void* const* d_in, const int* in_sizes, int n_in,
                              void* d_out, int out_size, void* d_ws, size_t ws_size,
                              hipStream_t stream) {
  (void)in_sizes; (void)n_in; (void)out_size; (void)ws_size;
  const float* x    = (const float*)d_in[0];
  const float* grid = (const float*)d_in[1];
  const float* bw   = (const float*)d_in[2];
  const float* sw   = (const float*)d_in[3];
  const float* sc   = (const float*)d_in[4];
  float*       out  = (float*)d_out;
  _Float16*    bfrag = (_Float16*)d_ws;   // 3*32*16 f16 = 3072 B

  kan_prep_weights<<<1, 256, 0, stream>>>(bw, sw, sc, bfrag);

  const int nblocks = (NIMG * TILES_PER_IMG) / WAVES;  // 30976/8 = 3872
  kan_conv_wmma<<<nblocks, 32 * WAVES, 0, stream>>>(x, grid, bfrag, out);
}
